// InferenceSuperNetwork_AttTra_AttRot_47012712022325
// MI455X (gfx1250) — matrix operand order repaired and checked by reading, not compiled, not run on hardware
//
#include <hip/hip_runtime.h>
#include <hip/hip_bf16.h>

typedef __attribute__((ext_vector_type(16))) _Float16 v16h;
typedef __attribute__((ext_vector_type(8)))  float    v8f;
typedef __attribute__((ext_vector_type(4)))  unsigned int v4u;

#define R_MAX 8
#define KN    128
#define NCH   1024     // KN * R_MAX output channels of the group conv
#define HW    2304     // 48*48
#define NB    8
#define TP17  320      // 9 paired-row chunks (16-wide) + 1 column chunk
#define TP33  1120     // 33 row chunks (32-wide) + 2 column chunks

// output section offsets (float elements)
#define OUT_ATTN  0
#define OUT_Q     147456
#define OUT_PR    294912
#define OUT_A     294920
#define OUT_OFF   442376
#define OUT_THETA 442384
#define OUT_Z     737296

__constant__ float c_offs[8] = {0.f, 0.78539816f, 1.57079633f, 2.35619449f,
                                3.14159265f, -2.35619449f, -1.57079633f, -0.78539816f};

__device__ __forceinline__ float lrelu(float v) { return v >= 0.f ? v : 0.01f * v; }

union AFrag { v4u q[2]; v16h h; };

// Storage-index -> (dy,dx,valid) mapping for the re-laid-out rotated weights.
// Chosen so that, per 32-tap chunk, the A-matrix patch offset is
// (per-lane base) + (compile-time immediate) for the CDNA5 16-bit A layout
// (lanes 0-15: K in {0..7,16..23}, lanes 16-31: K in {8..15,24..31}).
__device__ __forceinline__ void tap_map17(int s, int& dy, int& dx, bool& valid) {
  if (s < 288) {            // 9 chunks: two 16-wide half-rows per chunk
    int c = s >> 5, K = s & 31;
    dy = 2 * c + (K >> 4);
    dx = K & 15;
    valid = dy < 17;
  } else {                  // 1 chunk: column dx=16, dy = K
    int K = s - 288;
    dy = K; dx = 16;
    valid = K < 17;
  }
}
__device__ __forceinline__ void tap_map33(int s, int& dy, int& dx, bool& valid) {
  if (s < 1056) {           // 33 chunks: one 32-wide row each
    dy = s >> 5; dx = s & 31;
    valid = true;
  } else {                  // 2 chunks: column dx=32, dy = K
    int K = s - 1056;
    dy = K; dx = 32;
    valid = K < 33;
  }
}

// ---------------------------------------------------------------------------
// Kernel 1: rotate square kernels (bilinear, zero pad, align_corners=False).
// Writes CHANNEL-MAJOR f16 twT[c][s] using the chunk-friendly tap order above;
// invalid (padding) slots are zero so the GEMM needs no A-side tail guards.
// ---------------------------------------------------------------------------
__global__ void rotate_kernel(const float* __restrict__ w, _Float16* __restrict__ twT,
                              int ksize, int taps_pad) {
  int idx = blockIdx.x * blockDim.x + threadIdx.x;
  if (idx >= taps_pad * NCH) return;
  int c = idx / taps_pad;       // channel = o*8 + r
  int s = idx - c * taps_pad;   // storage tap index
  int o = c >> 3, r = c & 7;
  int dy, dx; bool valid;
  if (ksize == 17) tap_map17(s, dy, dx, valid);
  else             tap_map33(s, dy, dx, valid);
  float out = 0.f;
  if (valid) {
    int k = ksize, kk = k * k;
    float theta = 0.78539816339f * (float)r;   // 2*pi/8 * r
    float ct = cosf(theta), st = sinf(theta);
    float invk = 1.f / (float)k;
    float gy = (2.f * (float)dy + 1.f) * invk - 1.f;
    float gx = (2.f * (float)dx + 1.f) * invk - 1.f;
    float sx = ct * gx - st * gy;
    float sy = st * gx + ct * gy;
    float fx = ((sx + 1.f) * (float)k - 1.f) * 0.5f;
    float fy = ((sy + 1.f) * (float)k - 1.f) * 0.5f;
    int x0 = (int)floorf(fx), y0 = (int)floorf(fy);
    float wx1 = fx - (float)x0, wy1 = fy - (float)y0;
    const float* wo = w + o * kk;
    auto g = [&](int yy, int xx) -> float {
      if (yy < 0 || yy >= k || xx < 0 || xx >= k) return 0.f;
      return wo[yy * k + xx];
    };
    out = g(y0, x0)         * (1.f - wy1) * (1.f - wx1)
        + g(y0, x0 + 1)     * (1.f - wy1) * wx1
        + g(y0 + 1, x0)     * wy1 * (1.f - wx1)
        + g(y0 + 1, x0 + 1) * wy1 * wx1;
  }
  twT[idx] = (_Float16)out;
}

// ---------------------------------------------------------------------------
// Kernel 2: fused implicit-GEMM group conv (17x17 pad0 + 33x33 pad8) via WMMA.
// Block = 128 threads = 4 waves. Tile = 16 pixels (4x4) x 64 channels.
// All A elements: ds_load_u16 with immediate offset from 4 per-lane bases.
// B fragments: two 16B loads from channel-major weights. Zero tail guards
// (padding weight rows are zero; patch is staged finite everywhere reachable).
// ---------------------------------------------------------------------------
__global__ __launch_bounds__(128)
void groupconv_wmma(const float* __restrict__ x,
                    const _Float16* __restrict__ tw17,
                    const _Float16* __restrict__ tw33,
                    const float* __restrict__ b17,
                    const float* __restrict__ b33,
                    const float* __restrict__ ksw,
                    const float* __restrict__ gks,
                    _Float16* __restrict__ xout) {
  __shared__ _Float16 patch[44 * 36];   // 36x36 live + finite padding rows
  int tile = blockIdx.x;
  int b = tile / 144;
  int rem = tile % 144;
  int y0 = (rem / 12) * 4;
  int x0 = (rem % 12) * 4;
  int lane = threadIdx.x & 31;
  int wave = threadIdx.x >> 5;
  int c_base = blockIdx.y * 64 + wave * 16;

  const float* img = x + b * 4096;
  for (int i = threadIdx.x; i < 44 * 36; i += 128) {
    int pr = i / 36, pc = i % 36;
    int iy = y0 - 8 + pr, ix = x0 - 8 + pc;
    float v = (iy >= 0 && iy < 64 && ix >= 0 && ix < 64) ? img[iy * 64 + ix] : 0.f;
    patch[i] = (_Float16)v;
  }
  __syncthreads();

  int m = lane & 15;
  int mrow = m >> 2, mcol = m & 3;
  int ncol = lane & 15;
  int abase = (lane < 16) ? 0 : 8;     // A-fragment K group start
  int base16 = (lane < 16) ? 0 : 8;    // B-fragment K start

  // per-lane A bases (fold pixel position + abase so chunk offsets are immediates)
  const _Float16* base17 = &patch[(mrow + 8) * 36 + (mcol + 8) + abase];
  const _Float16* col17  = &patch[(mrow + 8 + abase) * 36 + (mcol + 8 + 16)];
  const _Float16* base33 = &patch[mrow * 36 + mcol + abase];
  const _Float16* col33  = &patch[(mrow + abase) * 36 + (mcol + 32)];
  const _Float16* brow17 = tw17 + (size_t)(c_base + ncol) * TP17 + base16;
  const _Float16* brow33 = tw33 + (size_t)(c_base + ncol) * TP33 + base16;

  // ---- 17x17, pad 0 ----
  v8f acc17 = {};
#pragma unroll
  for (int c = 0; c < 9; ++c) {        // paired 16-wide half-rows
    v16h a;
#pragma unroll
    for (int e = 0; e < 16; ++e) {
      int off = (e < 8) ? ((2 * c) * 36 + e) : ((2 * c + 1) * 36 + (e - 8));
      a[e] = base17[off];
    }
    AFrag bq; bq.q[0] = *(const v4u*)(brow17 + c * 32); bq.q[1] = *(const v4u*)(brow17 + c * 32 + 16); v16h bf = bq.h;
    acc17 = __builtin_amdgcn_wmma_f32_16x16x32_f16(false, a, false, bf,
                                                   (short)0, acc17, false, false);
  }
  {                                     // column dx=16 chunk (kb=288), dy=K
    v16h a;
#pragma unroll
    for (int e = 0; e < 16; ++e) {
      _Float16 av = (_Float16)0.f;     // K>16 rows have zero weights
      if (e <= 8) av = col17[((e < 8) ? e : e + 8) * 36];
      a[e] = av;
    }
    AFrag bq; bq.q[0] = *(const v4u*)(brow17 + 288); bq.q[1] = *(const v4u*)(brow17 + 288 + 16); v16h bf = bq.h;
    acc17 = __builtin_amdgcn_wmma_f32_16x16x32_f16(false, a, false, bf,
                                                   (short)0, acc17, false, false);
  }

  // ---- 33x33, pad 8 ----
  v8f acc33 = {};
#pragma unroll
  for (int c = 0; c < 33; ++c) {       // one 32-wide row per chunk
    v16h a;
#pragma unroll
    for (int e = 0; e < 16; ++e) {
      int off = c * 36 + ((e < 8) ? e : e + 8);
      a[e] = base33[off];
    }
    AFrag bq; bq.q[0] = *(const v4u*)(brow33 + c * 32); bq.q[1] = *(const v4u*)(brow33 + c * 32 + 16); v16h bf = bq.h;
    acc33 = __builtin_amdgcn_wmma_f32_16x16x32_f16(false, a, false, bf,
                                                   (short)0, acc33, false, false);
  }
  {                                     // column dx=32, chunk 1: dy = K (0..31)
    v16h a;
#pragma unroll
    for (int e = 0; e < 16; ++e) a[e] = col33[((e < 8) ? e : e + 8) * 36];
    AFrag bq; bq.q[0] = *(const v4u*)(brow33 + 1056); bq.q[1] = *(const v4u*)(brow33 + 1056 + 16); v16h bf = bq.h;
    acc33 = __builtin_amdgcn_wmma_f32_16x16x32_f16(false, a, false, bf,
                                                   (short)0, acc33, false, false);
  }
  {                                     // column dx=32, chunk 2: dy = 32+K (only K=0 live)
    v16h a;
#pragma unroll
    for (int e = 1; e < 16; ++e) a[e] = (_Float16)0.f;
    a[0] = col33[32 * 36];
    AFrag bq; bq.q[0] = *(const v4u*)(brow33 + 1088); bq.q[1] = *(const v4u*)(brow33 + 1088 + 16); v16h bf = bq.h;
    acc33 = __builtin_amdgcn_wmma_f32_16x16x32_f16(false, a, false, bf,
                                                   (short)0, acc33, false, false);
  }

  // alphas = softmax((ksize_weight + g_ksize)/tau)
  float l0 = (ksw[0] + gks[0]) * 0.1f;
  float l1 = (ksw[1] + gks[1]) * 0.1f;
  float mx = fmaxf(l0, l1);
  float e0 = expf(l0 - mx), e1 = expf(l1 - mx);
  float a0 = e0 / (e0 + e1), a1 = e1 / (e0 + e1);

  int c = c_base + ncol;
  int o = c >> 3, r = c & 7;
  float bias0 = b17[o], bias1 = b33[o];
#pragma unroll
  for (int v = 0; v < 8; ++v) {
    int mm = v + ((lane < 16) ? 0 : 8);
    int py = y0 + (mm >> 2), px = x0 + (mm & 3);
    float xo = a0 * lrelu(acc17[v] + bias0) + a1 * lrelu(acc33[v] + bias1);
    xout[(size_t)((b * 8 + r) * HW + py * 48 + px) * KN + o] = (_Float16)xo;
  }
}

// ---------------------------------------------------------------------------
// Kernel 3: conv2 (128x128 per-pixel GEMM via WMMA) + lrelu + conva/convz heads.
// ---------------------------------------------------------------------------
__global__ __launch_bounds__(256)
void head_wmma(const _Float16* __restrict__ xout,
               const float* __restrict__ w2, const float* __restrict__ b2,
               const float* __restrict__ wa, const float* __restrict__ ba,
               const float* __restrict__ wz, const float* __restrict__ bz,
               float* __restrict__ attn_o, float* __restrict__ z_o,
               int variant, int Rsub) {
  __shared__ _Float16 wBT[KN * KN];   // [n][k] == conv2_w[o][c]
  __shared__ float    hbuf[16 * KN];
  int tid = threadIdx.x;
  const float* wsrc = w2 + variant * KN * KN;
  for (int i = tid; i < KN * KN; i += 256) wBT[i] = (_Float16)wsrc[i];
  __syncthreads();

  int lane = tid & 31, wave = tid >> 5;
  int n_base = wave * 16;
  int ncol = lane & 15;
  int m = lane & 15;
  int p = blockIdx.x * 16 + m;
  int hw = p % HW;
  int rr = (p / HW) % Rsub;
  int b = p / (HW * Rsub);
  int r = (variant == 0) ? (rr * 2) : rr;
  const _Float16* arow = xout + (size_t)((b * 8 + r) * HW + hw) * KN;
  int ab = (lane < 16) ? 0 : 8;
  int base16 = (lane < 16) ? 0 : 8;
  const _Float16* brow = &wBT[(n_base + ncol) * KN + base16];

  v8f acc = {};
#pragma unroll
  for (int kb = 0; kb < KN; kb += 32) {
    AFrag af;
    af.q[0] = *(const v4u*)(arow + kb + ab);        // K = ab .. ab+7
    af.q[1] = *(const v4u*)(arow + kb + ab + 16);   // K = ab+16 .. ab+23
    AFrag bq; bq.q[0] = *(const v4u*)(brow + kb); bq.q[1] = *(const v4u*)(brow + kb + 16); v16h bf = bq.h;
    acc = __builtin_amdgcn_wmma_f32_16x16x32_f16(false, af.h, false, bf,
                                                 (short)0, acc, false, false);
  }
  float bias = b2[variant * KN + n_base + ncol];
#pragma unroll
  for (int v = 0; v < 8; ++v) {
    int mm = v + ((lane < 16) ? 0 : 8);
    hbuf[mm * KN + n_base + ncol] = lrelu(acc[v] + bias);
  }
  __syncthreads();

  if (tid < 80) {
    int pm = tid / 5, j = tid % 5;
    int pp = blockIdx.x * 16 + pm;
    const float* hrow = hbuf + pm * KN;
    if (j == 0) {
      float s = ba[variant];
      for (int o = 0; o < KN; ++o) s += wa[variant * KN + o] * hrow[o];
      attn_o[pp] = s;
    } else {
      int jj = j - 1;
      float s = bz[variant * 4 + jj];
      for (int o = 0; o < KN; ++o) s += wz[(variant * 4 + jj) * KN + o] * hrow[o];
      z_o[(size_t)pp * 4 + jj] = s;
    }
  }
}

// ---------------------------------------------------------------------------
// Kernel 4: betas, scatter-sum over R, + p_r; writes attn, z, theta, p_r, offsets
// ---------------------------------------------------------------------------
__global__ void combine_kernel(const float* __restrict__ attn0, const float* __restrict__ attn1,
                               const float* __restrict__ z0, const float* __restrict__ z1,
                               const float* __restrict__ rdw, const float* __restrict__ grd,
                               float* __restrict__ out) {
  int id = blockIdx.x * blockDim.x + threadIdx.x;
  if (id >= NB * R_MAX * HW) return;
  int hw = id % HW;
  int r = (id / HW) & 7;
  int b = id / (HW * 8);
  float l0 = (rdw[0] + grd[0]) * 0.1f;
  float l1 = (rdw[1] + grd[1]) * 0.1f;
  float mx = fmaxf(l0, l1);
  float e0 = expf(l0 - mx), e1 = expf(l1 - mx);
  float beta0 = e0 / (e0 + e1), beta1 = e1 / (e0 + e1);
  bool even = (r & 1) == 0;
  int p1 = (b * 8 + r) * HW + hw;
  int p0 = (b * 4 + (r >> 1)) * HW + hw;
  float off = c_offs[r];
  float t = off * 0.318309886f;                 // off / pi
  float pr = -0.5f * t * t - 1.14472989f - 0.91893853f;  // logN(0,pi) at off
  out[OUT_ATTN + id] = beta1 * attn1[p1] + (even ? beta0 * attn0[p0] : 0.f) + pr;
  for (int j = 0; j < 4; ++j) {
    float zv = beta1 * z1[(size_t)p1 * 4 + j] + (even ? beta0 * z0[(size_t)p0 * 4 + j] : 0.f);
    out[OUT_Z + ((b * 4 + j) * 8 + r) * HW + hw] = zv;
    if (j == 0) out[OUT_THETA + ((b * 2 + 0) * 8 + r) * HW + hw] = zv + off;
    if (j == 1) out[OUT_THETA + ((b * 2 + 1) * 8 + r) * HW + hw] = zv;
  }
  if (id < 8) {
    float o2 = c_offs[id];
    float t2 = o2 * 0.318309886f;
    out[OUT_PR + id]  = -0.5f * t2 * t2 - 1.14472989f - 0.91893853f;
    out[OUT_OFF + id] = o2;
  }
}

// ---------------------------------------------------------------------------
// Kernel 5a: per-batch max/sum stats for log_softmax (q) and gumbel softmax (a)
// ---------------------------------------------------------------------------
__global__ __launch_bounds__(256)
void stats_kernel(const float* __restrict__ attn, const float* __restrict__ gat,
                  float* __restrict__ stats) {
  __shared__ float red[256];
  int b = blockIdx.x, tid = threadIdx.x;
  const float* ab = attn + b * (R_MAX * HW);
  const float* gb = gat + b * (R_MAX * HW);
  float m1 = -1e30f, m2 = -1e30f;
  for (int i = tid; i < R_MAX * HW; i += 256) {
    float v = ab[i];
    m1 = fmaxf(m1, v);
    m2 = fmaxf(m2, v + gb[i]);
  }
  red[tid] = m1; __syncthreads();
  for (int s = 128; s > 0; s >>= 1) { if (tid < s) red[tid] = fmaxf(red[tid], red[tid + s]); __syncthreads(); }
  m1 = red[0]; __syncthreads();
  red[tid] = m2; __syncthreads();
  for (int s = 128; s > 0; s >>= 1) { if (tid < s) red[tid] = fmaxf(red[tid], red[tid + s]); __syncthreads(); }
  m2 = red[0]; __syncthreads();
  float s1 = 0.f, s2 = 0.f;
  for (int i = tid; i < R_MAX * HW; i += 256) {
    float v = ab[i];
    s1 += expf(v - m1);
    s2 += expf(v + gb[i] - m2);
  }
  red[tid] = s1; __syncthreads();
  for (int s = 128; s > 0; s >>= 1) { if (tid < s) red[tid] += red[tid + s]; __syncthreads(); }
  s1 = red[0]; __syncthreads();
  red[tid] = s2; __syncthreads();
  for (int s = 128; s > 0; s >>= 1) { if (tid < s) red[tid] += red[tid + s]; __syncthreads(); }
  if (tid == 0) { stats[b * 4 + 0] = m1; stats[b * 4 + 1] = s1; stats[b * 4 + 2] = m2; stats[b * 4 + 3] = red[0]; }
}

// ---------------------------------------------------------------------------
// Kernel 5b: q_t_r = log_softmax(attn); a_sampled = softmax(attn + g_attn)
// ---------------------------------------------------------------------------
__global__ void softmax_kernel(const float* __restrict__ gat, const float* __restrict__ stats,
                               float* __restrict__ out) {
  int id = blockIdx.x * blockDim.x + threadIdx.x;
  if (id >= NB * R_MAX * HW) return;
  int b = id / (R_MAX * HW);
  float m1 = stats[b * 4 + 0], s1 = stats[b * 4 + 1];
  float m2 = stats[b * 4 + 2], s2 = stats[b * 4 + 3];
  float v = out[OUT_ATTN + id];
  out[OUT_Q + id] = v - m1 - logf(s1);
  out[OUT_A + id] = expf(v + gat[id] - m2) / s2;
}

// ---------------------------------------------------------------------------
extern "C" void kernel_launch(void* const* d_in, const int* in_sizes, int n_in,
                              void* d_out, int out_size, void* d_ws, size_t ws_size,
                              hipStream_t stream) {
  const float* x   = (const float*)d_in[0];
  const float* ksw = (const float*)d_in[1];
  const float* rdw = (const float*)d_in[2];
  const float* w17 = (const float*)d_in[3];
  const float* b17 = (const float*)d_in[4];
  const float* w33 = (const float*)d_in[5];
  const float* b33 = (const float*)d_in[6];
  const float* w2  = (const float*)d_in[7];
  const float* b2  = (const float*)d_in[8];
  const float* wa  = (const float*)d_in[9];
  const float* ba  = (const float*)d_in[10];
  const float* wz  = (const float*)d_in[11];
  const float* bz  = (const float*)d_in[12];
  const float* gks = (const float*)d_in[13];
  const float* grd = (const float*)d_in[14];
  const float* gat = (const float*)d_in[15];
  float* out = (float*)d_out;

  // workspace layout (bytes, all 32B aligned)
  char* ws = (char*)d_ws;
  _Float16* tw17  = (_Float16*)(ws);                 // 1024*320 f16 =   655360
  _Float16* tw33  = (_Float16*)(ws + 655360);        // 1024*1120 f16 = 2293760
  _Float16* xout  = (_Float16*)(ws + 2949120);       // 8*8*2304*128 f16 = 37748736
  float*    attn0 = (float*)(ws + 40697856);         // 73728 f32
  float*    attn1 = (float*)(ws + 40992768);         // 147456 f32
  float*    z0    = (float*)(ws + 41582592);         // 73728*4 f32
  float*    z1    = (float*)(ws + 42762240);         // 147456*4 f32
  float*    stats = (float*)(ws + 45121536);         // 32 f32

  rotate_kernel<<<(TP17 * NCH + 255) / 256, 256, 0, stream>>>(w17, tw17, 17, TP17);
  rotate_kernel<<<(TP33 * NCH + 255) / 256, 256, 0, stream>>>(w33, tw33, 33, TP33);

  dim3 gconv(1152, 16);   // 1152 pixel tiles (8 batches * 12*12), 16 channel groups of 64
  groupconv_wmma<<<gconv, 128, 0, stream>>>(x, tw17, tw33, b17, b33, ksw, gks, xout);

  head_wmma<<<4608, 256, 0, stream>>>(xout, w2, b2, wa, ba, wz, bz, attn0, z0, 0, 4);
  head_wmma<<<9216, 256, 0, stream>>>(xout, w2, b2, wa, ba, wz, bz, attn1, z1, 1, 8);

  combine_kernel<<<(NB * R_MAX * HW + 255) / 256, 256, 0, stream>>>(attn0, attn1, z0, z1, rdw, grd, out);
  stats_kernel<<<NB, 256, 0, stream>>>(out + OUT_ATTN, gat, stats);
  softmax_kernel<<<(NB * R_MAX * HW + 255) / 256, 256, 0, stream>>>(gat, stats, out);
}
